// Attention_23493471109551
// MI455X (gfx1250) — matrix operand order repaired and checked by reading, not compiled, not run on hardware
//
#include <hip/hip_runtime.h>
#include <math.h>

// ---------------------------------------------------------------------------
// MI455X (gfx1250) attention pipeline, wave32 + v_wmma_f32_16x16x32_bf16.
//   B=4 S=2048 D=1024 H=16 Dh=64 ; scale = 1/sqrt(1024)
// Stages:
//   1) qp = rope(q @ Wq^T), kp = rope(k @ Wk^T)  -> bf16 [B,H,S,64]
//   2) vp = v @ Wv^T                             -> bf16 [B,H,S,64]
//   3) flash attention per (b,h), causal         -> bf16 [B,S,1024]
//   4) out = ao @ Wo^T                           -> f32  [B,S,1024]
// ---------------------------------------------------------------------------

typedef __attribute__((ext_vector_type(16))) __bf16 v16bf;
typedef __attribute__((ext_vector_type(8)))  float  v8f;
typedef int i32x4g __attribute__((vector_size(16)));   // matches async-LDS builtin

#define AS1 __attribute__((address_space(1)))
#define AS3 __attribute__((address_space(3)))

#if defined(__has_builtin)
#  if __has_builtin(__builtin_amdgcn_global_load_async_to_lds_b128) && \
      __has_builtin(__builtin_amdgcn_s_wait_asynccnt)
#    define HAVE_ASYNC_LDS 1
#  endif
#endif
#ifndef HAVE_ASYNC_LDS
#  define HAVE_ASYNC_LDS 0
#endif

union FragU {                 // 16 bf16 values = one WMMA A/B fragment per lane
    unsigned int u[8];
    v16bf        v;
};

__device__ __forceinline__ unsigned short f2bf(float f) {
    unsigned int u = __float_as_uint(f);
    unsigned int r = u + 0x7FFFu + ((u >> 16) & 1u);   // round-to-nearest-even
    return (unsigned short)(r >> 16);
}

#define C_B   4
#define C_S   2048
#define C_H   16
#define C_DH  64
#define C_D   1024
#define C_M   8192          // B*S

// ---------------------------------------------------------------------------
// Generic bf16-WMMA GEMM:  C[M,N] = A[M,K] * W[N,K]^T
//   KIND 0: A fp32, epilogue = RoPE + scatter bf16 to [B,H,S,64]   (q/k proj)
//   KIND 1: A fp32, epilogue = scatter bf16 to [B,H,S,64]          (v proj)
//   KIND 2: A bf16 (async->LDS), epilogue = fp32 row-major [M,N]   (out proj)
// Block tile 256(M) x 64(N), K-step 32, 256 threads = 8 wave32 waves.
// Wave w computes rows [w*32, w*32+32) x all 64 N columns:
//   2 A fragments, 4 B fragments (each reused twice) -> 8 WMMAs / K-step.
// ---------------------------------------------------------------------------
template<int KIND>
__global__ __launch_bounds__(256)
void gemm_bf16_wmma(const void* __restrict__ Aop, const float* __restrict__ W,
                    void* __restrict__ Oop, int Ntot, int Ktot)
{
    __shared__ __align__(16) unsigned short As[256 * 32];
    __shared__ __align__(16) unsigned short Bs[64 * 32];

    const int tid  = threadIdx.x;
    const int lane = tid & 31;
    const int wv   = tid >> 5;       // wave 0..7
    const int half = lane >> 4;      // 0: lanes 0-15, 1: lanes 16-31
    const int l16  = lane & 15;
    const int m0   = blockIdx.x * 256;
    const int n0   = blockIdx.y * 64;

    v8f acc[2][4] = {};

    for (int k0 = 0; k0 < Ktot; k0 += 32) {
        // ---- stage A tile (256 x 32) into LDS as bf16 ----
        if (KIND == 2) {
            const unsigned short* A = (const unsigned short*)Aop;
            #pragma unroll
            for (int i = 0; i < 4; ++i) {
                int f = tid + 256 * i;               // 1024 x 16B chunks
                int row = f >> 2, c8 = (f & 3) << 3;
#if HAVE_ASYNC_LDS
                __builtin_amdgcn_global_load_async_to_lds_b128(
                    (AS1 i32x4g*)(A + (size_t)(m0 + row) * Ktot + k0 + c8),
                    (AS3 i32x4g*)&As[row * 32 + c8], 0, 0);
#else
                *(uint4*)&As[row * 32 + c8] =
                    *(const uint4*)(A + (size_t)(m0 + row) * Ktot + k0 + c8);
#endif
            }
        } else {
            const float* A = (const float*)Aop;
            #pragma unroll
            for (int i = 0; i < 8; ++i) {
                int f = tid + 256 * i;               // 2048 x float4 chunks
                int row = f >> 3, c4 = (f & 7) << 2;
                float4 x = *(const float4*)(A + (size_t)(m0 + row) * Ktot + k0 + c4);
                ushort4 hh;
                hh.x = f2bf(x.x); hh.y = f2bf(x.y); hh.z = f2bf(x.z); hh.w = f2bf(x.w);
                *(ushort4*)&As[row * 32 + c4] = hh;
            }
        }
        // ---- stage B tile (64 x 32) = W rows n0..n0+63 ----
        #pragma unroll
        for (int i = 0; i < 2; ++i) {
            int f = tid + 256 * i;                   // 512 x float4 chunks
            int row = f >> 3, c4 = (f & 7) << 2;
            float4 x = *(const float4*)(W + (size_t)(n0 + row) * Ktot + k0 + c4);
            ushort4 hh;
            hh.x = f2bf(x.x); hh.y = f2bf(x.y); hh.z = f2bf(x.z); hh.w = f2bf(x.w);
            *(ushort4*)&Bs[row * 32 + c4] = hh;
        }
        if (k0 + 32 < Ktot) {                        // global_prefetch_b8 next B tile
            __builtin_prefetch(W + (size_t)(n0 + (tid >> 2)) * Ktot + k0 + 32, 0, 1);
        }
#if HAVE_ASYNC_LDS
        if (KIND == 2) __builtin_amdgcn_s_wait_asynccnt(0);
#endif
        __syncthreads();

        // ---- A fragments: 16x32, lane<16 holds K {0..7,16..23}, lane>=16 {8..15,24..31}
        FragU af[2];
        #pragma unroll
        for (int mi = 0; mi < 2; ++mi) {
            const unsigned short* ar = &As[(wv * 32 + mi * 16 + l16) * 32];
            ((uint4*)af[mi].u)[0] = *(const uint4*)(ar + half * 8);
            ((uint4*)af[mi].u)[1] = *(const uint4*)(ar + 16 + half * 8);
        }

        // ---- 4 B fragments (N columns frag*16 + l16), each reused for 2 WMMAs
        #pragma unroll
        for (int frag = 0; frag < 4; ++frag) {
            FragU bf_;
            const unsigned short* br = &Bs[(frag * 16 + l16) * 32 + half * 8];
            ((uint4*)bf_.u)[0] = *(const uint4*)(br);
            ((uint4*)bf_.u)[1] = *(const uint4*)(br + 16);
            acc[0][frag] = __builtin_amdgcn_wmma_f32_16x16x32_bf16(
                false, af[0].v, false, bf_.v, (short)0, acc[0][frag], false, false);
            acc[1][frag] = __builtin_amdgcn_wmma_f32_16x16x32_bf16(
                false, af[1].v, false, bf_.v, (short)0, acc[1][frag], false, false);
        }
        __syncthreads();
    }

    // ---- epilogue: C layout is M = v + 8*half, N = l16 (per fragment) ----
    #pragma unroll
    for (int mi = 0; mi < 2; ++mi) {
        if (KIND == 2) {
            float* O = (float*)Oop;
            #pragma unroll
            for (int frag = 0; frag < 4; ++frag)
                #pragma unroll
                for (int v = 0; v < 8; ++v) {
                    int mrow = m0 + wv * 32 + mi * 16 + v + 8 * half;
                    int n    = n0 + frag * 16 + l16;
                    O[(size_t)mrow * Ntot + n] = acc[mi][frag][v];
                }
        } else {
            unsigned short* O = (unsigned short*)Oop;
            #pragma unroll
            for (int frag = 0; frag < 4; ++frag)
                #pragma unroll
                for (int v = 0; v < 8; ++v) {
                    int mrow = m0 + wv * 32 + mi * 16 + v + 8 * half;
                    int n    = n0 + frag * 16 + l16;
                    int bb = mrow >> 11, ss = mrow & (C_S - 1);
                    int hh = n >> 6,     d  = n & 63;
                    float x = acc[mi][frag][v];
                    float o;
                    if (KIND == 0) {
                        // RoPE: pair (2i, 2i+1) sits on adjacent lanes -> shfl_xor(1)
                        float freq = __expf(-0.14391157f * (float)(d & ~1)); // ln(1e4)/64
                        float ang  = (float)ss * freq;
                        float sn, cs;
                        __sincosf(ang, &sn, &cs);
                        float p = __shfl_xor(x, 1);
                        o = (d & 1) ? fmaf(p, sn, x * cs)       // o2 = x1*sin + x2*cos
                                    : fmaf(x, cs, -(p * sn));   // o1 = x1*cos - x2*sin
                    } else {
                        o = x;
                    }
                    O[((size_t)(bb * C_H + hh) * C_S + ss) * C_DH + d] = f2bf(o);
                }
        }
    }
}

// ---------------------------------------------------------------------------
// Flash attention, causal. Grid: (S/128, H, B). 256 threads = 8 waves.
// Wave w owns query rows qb + [w*16, w*16+16). Key tiles of 64, K/V in LDS
// (K staged with async global->LDS when available; V stored transposed
// [d][key]); online softmax; P converted C->A layout via per-wave LDS
// scratch; two WMMA stages per tile (QK^T and PV).
// ---------------------------------------------------------------------------
__global__ __launch_bounds__(256)
void flash_attn_wmma(const unsigned short* __restrict__ qp,
                     const unsigned short* __restrict__ kp,
                     const unsigned short* __restrict__ vp,
                     unsigned short* __restrict__ ao)
{
    __shared__ __align__(16) unsigned short Ks[64 * 64];      // [key][d]
    __shared__ __align__(16) unsigned short Vs[64 * 64];      // [d][key] (transposed)
    __shared__ __align__(16) unsigned short Ps[8][16 * 64];   // per-wave P scratch

    const int tid  = threadIdx.x;
    const int lane = tid & 31;
    const int wv   = tid >> 5;
    const int half = lane >> 4;
    const int l16  = lane & 15;

    const int qb = blockIdx.x * 128;
    const int h  = blockIdx.y;
    const int b  = blockIdx.z;
    const size_t headOff = (size_t)(b * C_H + h) * C_S * C_DH;
    const unsigned short* Q = qp + headOff;
    const unsigned short* K = kp + headOff;
    const unsigned short* V = vp + headOff;

    // ---- Q fragments (A-layout): row = qb + wv*16 + l16, two K-steps of 32
    FragU aq[2];
    {
        const unsigned short* qr = Q + (size_t)(qb + wv * 16 + l16) * C_DH;
        #pragma unroll
        for (int kk = 0; kk < 2; ++kk) {
            ((uint4*)aq[kk].u)[0] = *(const uint4*)(qr + kk * 32 + half * 8);
            ((uint4*)aq[kk].u)[1] = *(const uint4*)(qr + kk * 32 + 16 + half * 8);
        }
    }

    float m_i[8], l_i[8];
    #pragma unroll
    for (int v = 0; v < 8; ++v) { m_i[v] = -__builtin_inff(); l_i[v] = 0.f; }
    v8f o_acc[4] = {};

    const float scale = 0.03125f;                  // 1/sqrt(1024)
    const int ntiles = (qb >> 6) + 2;              // keys up to qb+127

    for (int t = 0; t < ntiles; ++t) {
        const int kb = t * 64;
        // ---- stage K tile [64][64] (async DMA to LDS when available) ----
        #pragma unroll
        for (int i = 0; i < 2; ++i) {
            int f = tid + 256 * i;                 // 512 x 16B chunks
            int row = f >> 3, c8 = (f & 7) << 3;
#if HAVE_ASYNC_LDS
            __builtin_amdgcn_global_load_async_to_lds_b128(
                (AS1 i32x4g*)(K + (size_t)(kb + row) * C_DH + c8),
                (AS3 i32x4g*)&Ks[row * 64 + c8], 0, 0);
#else
            *(uint4*)&Ks[row * 64 + c8] =
                *(const uint4*)(K + (size_t)(kb + row) * C_DH + c8);
#endif
        }
        // ---- stage V tile transposed: Vs[d][key] ----
        #pragma unroll
        for (int i = 0; i < 2; ++i) {
            int f = tid + 256 * i;
            int key = f >> 3, d0 = (f & 7) << 3;
            uint4 raw = *(const uint4*)(V + (size_t)(kb + key) * C_DH + d0);
            const unsigned short* e = (const unsigned short*)&raw;
            #pragma unroll
            for (int j = 0; j < 8; ++j) Vs[(d0 + j) * 64 + key] = e[j];
        }
        if (t + 1 < ntiles) {
            __builtin_prefetch(K + (size_t)(kb + 64) * C_DH + tid * 16, 0, 1);
            __builtin_prefetch(V + (size_t)(kb + 64) * C_DH + tid * 16, 0, 1);
        }
#if HAVE_ASYNC_LDS
        __builtin_amdgcn_s_wait_asynccnt(0);
#endif
        __syncthreads();

        // ---- scores S = Q K^T : 8 WMMAs ----
        v8f sacc[4] = {};
        #pragma unroll
        for (int kk = 0; kk < 2; ++kk) {
            #pragma unroll
            for (int frag = 0; frag < 4; ++frag) {
                FragU bk;
                const unsigned short* br = &Ks[(frag * 16 + l16) * 64 + kk * 32 + half * 8];
                ((uint4*)bk.u)[0] = *(const uint4*)(br);
                ((uint4*)bk.u)[1] = *(const uint4*)(br + 16);
                sacc[frag] = __builtin_amdgcn_wmma_f32_16x16x32_bf16(
                    false, aq[kk].v, false, bk.v, (short)0, sacc[frag], false, false);
            }
        }

        // ---- online softmax (row = qb + wv*16 + v + 8*half, key = kb + frag*16 + l16)
        #pragma unroll
        for (int v = 0; v < 8; ++v) {
            const int qr = qb + wv * 16 + v + 8 * half;
            float sv[4];
            float mx = m_i[v];
            #pragma unroll
            for (int frag = 0; frag < 4; ++frag) {
                int key = kb + frag * 16 + l16;
                float s = sacc[frag][v] * scale;
                if (key > qr) s = -__builtin_inff();   // causal mask
                sv[frag] = s;
                mx = fmaxf(mx, s);
            }
            #pragma unroll
            for (int off = 1; off < 16; off <<= 1) mx = fmaxf(mx, __shfl_xor(mx, off));
            const float fac = __expf(m_i[v] - mx);
            m_i[v] = mx;
            float rs = 0.f;
            #pragma unroll
            for (int frag = 0; frag < 4; ++frag) {
                float p = __expf(sv[frag] - mx);
                rs += p;
                Ps[wv][(v + 8 * half) * 64 + frag * 16 + l16] = f2bf(p);
            }
            #pragma unroll
            for (int off = 1; off < 16; off <<= 1) rs += __shfl_xor(rs, off);
            l_i[v] = l_i[v] * fac + rs;
            #pragma unroll
            for (int frag = 0; frag < 4; ++frag) o_acc[frag][v] *= fac;
        }

        // ---- O += P V : reload P in A-layout from per-wave scratch, 8 WMMAs ----
        #pragma unroll
        for (int kk = 0; kk < 2; ++kk) {
            FragU ap;
            const unsigned short* pr = &Ps[wv][l16 * 64 + kk * 32];
            ((uint4*)ap.u)[0] = *(const uint4*)(pr + half * 8);
            ((uint4*)ap.u)[1] = *(const uint4*)(pr + 16 + half * 8);
            #pragma unroll
            for (int frag = 0; frag < 4; ++frag) {
                FragU bv;
                const unsigned short* vr = &Vs[(frag * 16 + l16) * 64 + kk * 32 + half * 8];
                ((uint4*)bv.u)[0] = *(const uint4*)(vr);
                ((uint4*)bv.u)[1] = *(const uint4*)(vr + 16);
                o_acc[frag] = __builtin_amdgcn_wmma_f32_16x16x32_bf16(
                    false, ap.v, false, bv.v, (short)0, o_acc[frag], false, false);
            }
        }
        __syncthreads();
    }

    // ---- normalize + store bf16 to [B,S,1024] (A-layout for final GEMM) ----
    #pragma unroll
    for (int frag = 0; frag < 4; ++frag)
        #pragma unroll
        for (int v = 0; v < 8; ++v) {
            int qr = qb + wv * 16 + v + 8 * half;
            int d  = frag * 16 + l16;
            float o = o_acc[frag][v] / l_i[v];
            ao[(size_t)(b * C_S + qr) * C_D + h * C_DH + d] = f2bf(o);
        }
}

// ---------------------------------------------------------------------------
extern "C" void kernel_launch(void* const* d_in, const int* in_sizes, int n_in,
                              void* d_out, int out_size, void* d_ws, size_t ws_size,
                              hipStream_t stream) {
    const float* q   = (const float*)d_in[0];
    const float* k   = (const float*)d_in[1];
    const float* v   = (const float*)d_in[2];
    const float* W_q = (const float*)d_in[3];
    const float* W_k = (const float*)d_in[4];
    const float* W_v = (const float*)d_in[5];
    const float* W_o = (const float*)d_in[6];

    const size_t nElem = (size_t)C_M * C_D;          // 8M elems per buffer
    unsigned short* qp = (unsigned short*)d_ws;      // bf16 [B,H,S,64]
    unsigned short* kp = qp + nElem;
    unsigned short* vp = kp + nElem;
    unsigned short* ao = vp + nElem;                 // bf16 [B,S,1024]

    dim3 blk(256);
    dim3 gG(C_M / 256, C_D / 64);                    // 32 x 16 blocks
    gemm_bf16_wmma<0><<<gG, blk, 0, stream>>>((const void*)q, W_q, (void*)qp, C_D, 2048);
    gemm_bf16_wmma<0><<<gG, blk, 0, stream>>>((const void*)k, W_k, (void*)kp, C_D, 2048);
    gemm_bf16_wmma<1><<<gG, blk, 0, stream>>>((const void*)v, W_v, (void*)vp, C_D, 1024);

    dim3 gF(C_S / 128, C_H, C_B);                    // 16 x 16 x 4 blocks
    flash_attn_wmma<<<gF, blk, 0, stream>>>(qp, kp, vp, ao);

    gemm_bf16_wmma<2><<<gG, blk, 0, stream>>>((const void*)ao, W_o, d_out, C_D, 1024);
}
